// Decoder1_80590766342729
// MI455X (gfx1250) — matrix operand order repaired and checked by reading, not compiled, not run on hardware
//
#include <hip/hip_runtime.h>
#include <hip/hip_bf16.h>

typedef __attribute__((ext_vector_type(16))) _Float16 v16h;
typedef __attribute__((ext_vector_type(8)))  float    v8f;

#define B_TOT    1024
#define N_PTS    101
#define D_MOD    128
#define NH       8
#define HEAD_D   16
#define LDA      132      // padded f32 row stride (avoids 64-bank conflicts at stride 128)
#define NPAD     112
#define ROWS_TOT (B_TOT * N_PTS)

// ---------------- WMMA helpers (CDNA5 16x16x32 f16 -> f32) ----------------

__device__ __forceinline__ v8f wmma_f16(v16h a, v16h b, v8f c) {
  return __builtin_amdgcn_wmma_f32_16x16x32_f16(false, a, false, b, (short)0, c,
                                                false, false);
}

// B operand pre-packed in ISA 32x16 lane order: element e of lane L in tile
// (ct,kt) is  W[kt*32 + 8*(L/16) + e + 8*(e/8)][ct*16 + L%16]  -> one 32B contiguous load per lane.
__device__ __forceinline__ v16h load_b_packed(const _Float16* __restrict__ pk,
                                              int ct, int kt, int lane) {
  return *reinterpret_cast<const v16h*>(pk + ((size_t)((ct * 4 + kt) * 32 + lane) << 4));
}

// A operand gather from f32 row-major 16 x 128 tile per ISA 16x32 f16 A layout:
// lane<16: M=lane, K base 0/16 ; lane>=16: same M, K offset +8.
__device__ __forceinline__ v16h gather_a(const float* __restrict__ base,
                                         int stride, int lane, int k0) {
  const int m    = lane & 15;
  const int koff = (lane >= 16) ? 8 : 0;
  const float* row = base + (size_t)m * stride + k0 + koff;
  v16h a;
#pragma unroll
  for (int r = 0; r < 8; ++r) {
    const int kb = ((r < 4) ? 0 : 16) + 2 * (r & 3);
    a[2 * r]     = (_Float16)row[kb];
    a[2 * r + 1] = (_Float16)row[kb + 1];
  }
  return a;
}

// ---------------- wave32 reductions ----------------

__device__ __forceinline__ float wave_sum(float v) {
#pragma unroll
  for (int s = 16; s >= 1; s >>= 1) v += __shfl_xor(v, s, 32);
  return v;
}
__device__ __forceinline__ float wave_max(float v) {
#pragma unroll
  for (int s = 16; s >= 1; s >>= 1) v = fmaxf(v, __shfl_xor(v, s, 32));
  return v;
}

// ---------------- weight repack: f32 (128x128 row-major) -> WMMA-B f16 ----------------

__global__ __launch_bounds__(256) void repack_weight(const float* __restrict__ src,
                                                     _Float16* __restrict__ dst) {
  const int tid  = blockIdx.x * 256 + threadIdx.x;  // 0..1023 = ct*128 + kt*32 + lane
  const int lane = tid & 31;
  const int kt   = (tid >> 5) & 3;
  const int ct   = tid >> 7;
  const int K    = kt * 32 + ((lane >> 4) << 3);
#pragma unroll
  for (int e = 0; e < 16; ++e)
    dst[((size_t)tid << 4) + e] = (_Float16)src[(size_t)(K + e + ((e >> 3) << 3)) * D_MOD + ct * 16 + (lane & 15)];
}

// ---------------- precompute k, v, kp in f16 (L2-resident, half traffic) ----------------

__global__ __launch_bounds__(256) void kvp_precompute(
    const float* __restrict__ enc,
    const _Float16* __restrict__ pkk, const _Float16* __restrict__ pkv,
    const _Float16* __restrict__ pkp,
    _Float16* __restrict__ kf, _Float16* __restrict__ vf, _Float16* __restrict__ kpf) {
  __shared__ float s_a[16 * LDA];
  const int tid = threadIdx.x;
  const int r0  = blockIdx.x * 16;  // 16 rows of flattened (B*N)
  {
    const int row = tid >> 4, c0 = (tid & 15) * 8;
    const float* sp = enc + (size_t)(r0 + row) * D_MOD + c0;
#pragma unroll
    for (int e = 0; e < 8; ++e) s_a[row * LDA + c0 + e] = sp[e];
  }
  __syncthreads();
  const int wave = tid >> 5, lane = tid & 31;
  v8f ak = {}, av = {}, ap = {};
#pragma unroll
  for (int kt = 0; kt < 4; ++kt) {
    v16h a = gather_a(s_a, LDA, lane, kt * 32);
    ak = wmma_f16(a, load_b_packed(pkk, wave, kt, lane), ak);
    av = wmma_f16(a, load_b_packed(pkv, wave, kt, lane), av);
    ap = wmma_f16(a, load_b_packed(pkp, wave, kt, lane), ap);
  }
  const int lc = lane & 15, rad = (lane >= 16) ? 8 : 0;
  const int cg = wave * 16 + lc;
  const int h = cg >> 4, d = cg & 15;
#pragma unroll
  for (int r = 0; r < 8; ++r) {
    const int gr = r0 + r + rad;         // ROWS_TOT divisible by 16 -> always valid
    const int b  = gr / N_PTS;
    const int n  = gr - b * N_PTS;
    kf[((size_t)(b * NH + h) * N_PTS + n) * HEAD_D + d] = (_Float16)ak[r];
    vf[((size_t)(b * NH + h) * N_PTS + n) * HEAD_D + d] = (_Float16)av[r];
    kpf[((size_t)b * N_PTS + n) * D_MOD + cg]           = (_Float16)ap[r];
  }
}

// ---------------- mask update (reference _update_mask), one wave per sample ----------------

__device__ __forceinline__ void update_mask_wave(
    int bl, int b, int idx, float dyn, int nd,
    const float* __restrict__ demand, int lane,
    float (&s_mask)[16][NPAD], float (&s_mask1)[16][NPAD]) {
  const bool godep = idx < nd;
  if (!godep && lane == 0) s_mask1[bl][idx] = 1.0f;  // one-hot max (same-wave DS in-order)
  float s1 = 0.f;
  for (int n = lane; n < N_PTS; n += 32)
    if (n >= nd) s1 += s_mask1[bl][n];
  s1 = wave_sum(s1);
  const bool  done = s1 >= (float)(N_PTS - nd);
  const float dep  = (godep && !done) ? 1.0f : 0.0f;
  for (int n = lane; n < NPAD; n += 32) {
    float m;
    if (n < N_PTS) {
      m = fmaxf(s_mask1[bl][n], (demand[(size_t)b * N_PTS + n] > dyn) ? 1.0f : 0.0f);
      if (n < nd) m = dep;
    } else {
      m = 1.0f;  // padding permanently masked
    }
    s_mask[bl][n] = m;
  }
}

// ---------------- sequential decode: 16 samples per block, whole loop on-chip ----------------

__global__ __launch_bounds__(256) void vrp_decode(
    const float* __restrict__ enc, const float* __restrict__ pool,
    const float* __restrict__ cap, const float* __restrict__ demand,
    const float* __restrict__ fc_w_raw,
    const _Float16* __restrict__ pk_fc, const _Float16* __restrict__ pk_fc1,
    const _Float16* __restrict__ pk_aw, const _Float16* __restrict__ pk_afc,
    const _Float16* __restrict__ kf, const _Float16* __restrict__ vf,
    const _Float16* __restrict__ kpf,
    const int* __restrict__ p_steps, const int* __restrict__ p_nd,
    const int* __restrict__ p_temp,
    float* __restrict__ out) {
  __shared__ float s_state[16 * LDA], s_decin[16 * LDA], s_q[16 * LDA];
  __shared__ float s_att[16 * LDA], s_aout[16 * LDA], s_pool[16 * LDA];
  __shared__ float s_sc[8][NH * NPAD];  // per-wave softmax scores
  __shared__ float s_mask[16][NPAD], s_mask1[16][NPAD];
  __shared__ float s_dyn[16], s_logp[16];

  const int tid = threadIdx.x, wave = tid >> 5, lane = tid & 31;
  const int bb = blockIdx.x * 16;
  const int n_steps = *p_steps;
  const int nd = *p_nd;
  const float tinv = 1.0f / (float)(*p_temp);
  const float cap_full = cap[0];
  const float inv_hd = 0.25f;                   // 1/sqrt(16)
  const float inv_d  = 0.08838834764831845f;    // 1/sqrt(128)

  // ---- init: state = enc[:,0,:], masks = 0, dyn = capacity ----
  {
    const int row = tid >> 4, c0 = (tid & 15) * 8;
    const float* sp = enc + (size_t)(bb + row) * N_PTS * D_MOD + c0;
#pragma unroll
    for (int e = 0; e < 8; ++e) s_state[row * LDA + c0 + e] = sp[e];
    for (int n = tid & 15; n < NPAD; n += 16) { s_mask[row][n] = 0.f; s_mask1[row][n] = 0.f; }
  }
  if (tid < 16) { s_dyn[tid] = cap[bb + tid]; s_logp[tid] = 0.f; }
  __syncthreads();

  // ---- pooled = pool @ fc1_w (WMMA, batched over 16 samples) ----
  {
    v8f acc = {};
#pragma unroll
    for (int kt = 0; kt < 4; ++kt) {
      v16h a = gather_a(pool + (size_t)bb * D_MOD, D_MOD, lane, kt * 32);
      acc = wmma_f16(a, load_b_packed(pk_fc1, wave, kt, lane), acc);
    }
    const int lc = lane & 15, rad = (lane >= 16) ? 8 : 0, col = wave * 16 + lc;
#pragma unroll
    for (int r = 0; r < 8; ++r) s_pool[(r + rad) * LDA + col] = acc[r];
  }
  // ---- initial _update_mask(index=0) ----
#pragma unroll
  for (int sub = 0; sub < 2; ++sub) {
    const int bl = wave * 2 + sub;
    update_mask_wave(bl, bb + bl, 0, s_dyn[bl], nd, demand, lane, s_mask, s_mask1);
  }
  __syncthreads();

  for (int t = 0; t < n_steps; ++t) {
    const int lc = lane & 15, rad = (lane >= 16) ? 8 : 0, col = wave * 16 + lc;

    // ---- Phase 1: dec_in = [state|dyn] @ fc_w + pooled (WMMA + rank-1 dyn term) ----
    {
      v8f acc = {};
#pragma unroll
      for (int kt = 0; kt < 4; ++kt)
        acc = wmma_f16(gather_a(s_state, LDA, lane, kt * 32),
                       load_b_packed(pk_fc, wave, kt, lane), acc);
      const float wlast = fc_w_raw[(size_t)D_MOD * D_MOD + col];  // fc_w row 128
#pragma unroll
      for (int r = 0; r < 8; ++r) {
        const int row = r + rad;
        s_decin[row * LDA + col] = acc[r] + s_pool[row * LDA + col] + s_dyn[row] * wlast;
      }
    }
    __syncthreads();

    // ---- Phase 2: q = dec_in @ attn_w (WMMA) ----
    {
      v8f acc = {};
#pragma unroll
      for (int kt = 0; kt < 4; ++kt)
        acc = wmma_f16(gather_a(s_decin, LDA, lane, kt * 32),
                       load_b_packed(pk_aw, wave, kt, lane), acc);
#pragma unroll
      for (int r = 0; r < 8; ++r) s_q[(r + rad) * LDA + col] = acc[r];
    }
    __syncthreads();

    // ---- Phase 3: per-sample masked MHA (VALU; both operands batch-indexed) ----
    for (int sub = 0; sub < 2; ++sub) {
      const int bl = wave * 2 + sub;
      const int b  = bb + bl;
      float cm[4][NH];
#pragma unroll
      for (int i = 0; i < 4; ++i) {
        const int n = lane + 32 * i;
#pragma unroll
        for (int h = 0; h < NH; ++h) cm[i][h] = -3.0e38f;
        if (n < N_PTS) {
          const bool masked = s_mask[bl][n] > 0.f;
#pragma unroll
          for (int h = 0; h < NH; ++h) {
            const _Float16* kr = kf + ((size_t)(b * NH + h) * N_PTS + n) * HEAD_D;
            float a2 = 0.f;
#pragma unroll
            for (int d = 0; d < HEAD_D; ++d)
              a2 += (float)kr[d] * s_q[bl * LDA + h * HEAD_D + d];
            cm[i][h] = masked ? -1.0e9f : a2 * inv_hd;
          }
        }
      }
      float* sw = s_sc[wave];
#pragma unroll
      for (int h = 0; h < NH; ++h) {
        float mx = fmaxf(fmaxf(cm[0][h], cm[1][h]), fmaxf(cm[2][h], cm[3][h]));
        mx = wave_max(mx);
        float ex[4], sm = 0.f;
#pragma unroll
        for (int i = 0; i < 4; ++i) { ex[i] = __expf(cm[i][h] - mx); sm += ex[i]; }
        sm = wave_sum(sm);
        const float inv = 1.0f / sm;
#pragma unroll
        for (int i = 0; i < 4; ++i) {
          const int n = lane + 32 * i;
          if (n < NPAD) sw[h * NPAD + n] = ex[i] * inv;
        }
      }
      // out_heads = scores @ v
#pragma unroll
      for (int j = 0; j < 4; ++j) {
        const int idx2 = lane + 32 * j;
        const int h = idx2 >> 4, d = idx2 & 15;
        const _Float16* vb = vf + (size_t)(b * NH + h) * N_PTS * HEAD_D + d;
        float a2 = 0.f;
        for (int n = 0; n < N_PTS; ++n)
          a2 += sw[h * NPAD + n] * (float)vb[(size_t)n * HEAD_D];
        s_att[bl * LDA + idx2] = a2;
      }
    }
    __syncthreads();

    // ---- Phase 4: out = att @ attn_fc (WMMA) ----
    {
      v8f acc = {};
#pragma unroll
      for (int kt = 0; kt < 4; ++kt)
        acc = wmma_f16(gather_a(s_att, LDA, lane, kt * 32),
                       load_b_packed(pk_afc, wave, kt, lane), acc);
#pragma unroll
      for (int r = 0; r < 8; ++r) s_aout[(r + rad) * LDA + col] = acc[r];
    }
    __syncthreads();

    // ---- Phase 5: logits, argmax, log-softmax, state/mask/dyn update ----
    for (int sub = 0; sub < 2; ++sub) {
      const int bl = wave * 2 + sub;
      const int b  = bb + bl;
      float lg[4];
#pragma unroll
      for (int i = 0; i < 4; ++i) {
        const int n = lane + 32 * i;
        if (n < N_PTS) {
          const _Float16* kr = kpf + ((size_t)b * N_PTS + n) * D_MOD;
          float a2 = 0.f;
#pragma unroll 8
          for (int d = 0; d < D_MOD; ++d) a2 += s_aout[bl * LDA + d] * (float)kr[d];
          float l = tanhf(a2 * inv_d) * 10.0f;
          if (s_mask[bl][n] > 0.f) l = -1.0e9f;
          lg[i] = l * tinv;
        } else {
          lg[i] = -3.0e38f;
        }
      }
      // argmax with first-index tie-break (matches jnp.argmax)
      float bv = lg[0]; int bi = lane;
#pragma unroll
      for (int i = 1; i < 4; ++i) {
        const int n = lane + 32 * i;
        if (lg[i] > bv) { bv = lg[i]; bi = n; }
      }
#pragma unroll
      for (int s = 16; s >= 1; s >>= 1) {
        const float ov = __shfl_xor(bv, s, 32);
        const int   oi = __shfl_xor(bi, s, 32);
        if (ov > bv || (ov == bv && oi < bi)) { bv = ov; bi = oi; }
      }
      // log_softmax at argmax = -log(sum exp(l - max))
      float sme = 0.f;
#pragma unroll
      for (int i = 0; i < 4; ++i)
        if (lane + 32 * i < N_PTS) sme += __expf(lg[i] - bv);
      sme = wave_sum(sme);
      float logp = -__logf(sme);
      // is_done from pre-update mask1[:,1:]
      float s1 = 0.f;
      for (int n = lane; n < N_PTS; n += 32)
        if (n >= 1) s1 += s_mask1[bl][n];
      s1 = wave_sum(s1);
      if (s1 >= (float)(N_PTS - 1)) logp = 0.f;

      const float dynold = s_dyn[bl];
      const float seld   = demand[(size_t)b * N_PTS + bi];
      const float dynnew = (bi < nd) ? cap_full : (dynold - seld);
      if (lane == 0) {
        s_dyn[bl] = dynnew;
        s_logp[bl] += logp;
        out[(size_t)b * n_steps + t] = (float)bi;  // actions.T[b][t]
      }
      update_mask_wave(bl, b, bi, dynnew, nd, demand, lane, s_mask, s_mask1);
      // state = enc[b, index, :]
      const float* srow = enc + ((size_t)b * N_PTS + bi) * D_MOD;
      for (int d2 = lane; d2 < D_MOD; d2 += 32) s_state[bl * LDA + d2] = srow[d2];
    }
    __syncthreads();
  }

  if (tid < 16) out[(size_t)B_TOT * n_steps + bb + tid] = s_logp[tid];
}

// ---------------- host launch ----------------

extern "C" void kernel_launch(void* const* d_in, const int* in_sizes, int n_in,
                              void* d_out, int out_size, void* d_ws, size_t ws_size,
                              hipStream_t stream) {
  (void)in_sizes; (void)n_in; (void)out_size; (void)ws_size;
  const float* enc     = (const float*)d_in[0];
  const float* pool    = (const float*)d_in[1];
  const float* cap     = (const float*)d_in[2];
  const float* demand  = (const float*)d_in[3];
  const float* fc_w    = (const float*)d_in[4];
  const float* fc1_w   = (const float*)d_in[5];
  const float* attn_w  = (const float*)d_in[6];
  const float* attn_k  = (const float*)d_in[7];
  const float* attn_v  = (const float*)d_in[8];
  const float* attn_fc = (const float*)d_in[9];
  const float* prob_k  = (const float*)d_in[10];
  const int* n_steps   = (const int*)d_in[11];
  const int* nd        = (const int*)d_in[12];
  const int* temp      = (const int*)d_in[13];

  _Float16* ws = (_Float16*)d_ws;
  const size_t KVN = (size_t)B_TOT * NH * N_PTS * HEAD_D;  // == B*N*D elements
  _Float16* kf  = ws;
  _Float16* vf  = kf + KVN;
  _Float16* kpf = vf + KVN;
  _Float16* pk  = kpf + KVN;  // 7 packed 128x128 f16 weight images
  _Float16* pk_fc  = pk + 0 * 16384;
  _Float16* pk_fc1 = pk + 1 * 16384;
  _Float16* pk_aw  = pk + 2 * 16384;
  _Float16* pk_ak  = pk + 3 * 16384;
  _Float16* pk_av  = pk + 4 * 16384;
  _Float16* pk_afc = pk + 5 * 16384;
  _Float16* pk_pk  = pk + 6 * 16384;

  repack_weight<<<4, 256, 0, stream>>>(fc_w,    pk_fc);   // rows 0..127 of (129,128)
  repack_weight<<<4, 256, 0, stream>>>(fc1_w,   pk_fc1);
  repack_weight<<<4, 256, 0, stream>>>(attn_w,  pk_aw);
  repack_weight<<<4, 256, 0, stream>>>(attn_k,  pk_ak);
  repack_weight<<<4, 256, 0, stream>>>(attn_v,  pk_av);
  repack_weight<<<4, 256, 0, stream>>>(attn_fc, pk_afc);
  repack_weight<<<4, 256, 0, stream>>>(prob_k,  pk_pk);

  kvp_precompute<<<ROWS_TOT / 16, 256, 0, stream>>>(enc, pk_ak, pk_av, pk_pk, kf, vf, kpf);

  vrp_decode<<<B_TOT / 16, 256, 0, stream>>>(enc, pool, cap, demand, fc_w,
                                             pk_fc, pk_fc1, pk_aw, pk_afc,
                                             kf, vf, kpf, n_steps, nd, temp,
                                             (float*)d_out);
}
